// GATConv_SGRACE_70222715289885
// MI455X (gfx1250) — matrix operand order repaired and checked by reading, not compiled, not run on hardware
//
#include <hip/hip_runtime.h>
#include <hip/hip_bf16.h>

typedef __attribute__((ext_vector_type(16))) __bf16 v16bf;
typedef __attribute__((ext_vector_type(8)))  float  v8f;
typedef __attribute__((ext_vector_type(4)))  int    v4i;

#define N_NODES 8192
#define F_IN    512
#define F_OUT   64
#define ALPHA   0.2f
#define NEG_BIG -9.0e15f
#define NCHUNK  4
#define CHUNK_COLS (N_NODES / NCHUNK)   // 2048 columns per partial

union ABfrag { unsigned int u[8]; v16bf v; };

__device__ __forceinline__ unsigned int bfbits(float f) {
    unsigned int u = __float_as_uint(f);
    return (u + 0x7FFFu + ((u >> 16) & 1u)) >> 16;
}

// Pack two floats to packed bf16 pair: round-half-up bias then one v_perm_b32
__device__ __forceinline__ unsigned int packbf2(float lo, float hi) {
    unsigned int ul = __float_as_uint(lo) + 0x8000u;
    unsigned int uh = __float_as_uint(hi) + 0x8000u;
    return __builtin_amdgcn_perm(uh, ul, 0x07060302u);
}

// ---------------------------------------------------------------------------
// Kernel 0: WT[n][k] = bf16(W[k][n])   (64 x 512 bf16, B-operand friendly)
// ---------------------------------------------------------------------------
__global__ void gat_wt_kernel(const float* __restrict__ W,
                              unsigned short* __restrict__ WT) {
    int tid = blockIdx.x * blockDim.x + threadIdx.x;  // 0 .. 32767
    int k = tid >> 6;        // 0..511
    int n = tid & 63;        // 0..63
    WT[n * F_IN + k] = (unsigned short)bfbits(W[k * F_OUT + n]);
}

// ---------------------------------------------------------------------------
// Kernel 1: Wh = X @ W via v_wmma_f32_16x16x32_bf16.
// ---------------------------------------------------------------------------
__global__ void __launch_bounds__(256)
gat_proj_kernel(const float* __restrict__ X,
                const unsigned short* __restrict__ WT,
                float* __restrict__ Wh,
                unsigned short* __restrict__ WhT) {
    const int wave = threadIdx.x >> 5;
    const int lane = threadIdx.x & 31;
    const int lm   = lane & 15;
    const int lhi  = lane >> 4;
    const int rowbase = blockIdx.x * 128 + wave * 16;
    const int rowA = rowbase + lm;

    v8f acc0 = {}, acc1 = {}, acc2 = {}, acc3 = {};
    const float* xrow = X + (size_t)rowA * F_IN;

    #pragma unroll 4
    for (int kstep = 0; kstep < 16; ++kstep) {
        const int k0 = kstep * 32;
        const int kb = lhi * 16;

        ABfrag B0, B1, B2, B3;
        {
            const unsigned short* bp0 = WT + (0 * 16 + lm) * F_IN + k0 + lhi * 16;
            const unsigned short* bp1 = WT + (1 * 16 + lm) * F_IN + k0 + lhi * 16;
            const unsigned short* bp2 = WT + (2 * 16 + lm) * F_IN + k0 + lhi * 16;
            const unsigned short* bp3 = WT + (3 * 16 + lm) * F_IN + k0 + lhi * 16;
            *(uint4*)&B0.u[0] = *(const uint4*)(bp0);
            *(uint4*)&B0.u[4] = *(const uint4*)(bp0 + 8);
            *(uint4*)&B1.u[0] = *(const uint4*)(bp1);
            *(uint4*)&B1.u[4] = *(const uint4*)(bp1 + 8);
            *(uint4*)&B2.u[0] = *(const uint4*)(bp2);
            *(uint4*)&B2.u[4] = *(const uint4*)(bp2 + 8);
            *(uint4*)&B3.u[0] = *(const uint4*)(bp3);
            *(uint4*)&B3.u[4] = *(const uint4*)(bp3 + 8);
        }

        float4 xa0 = *(const float4*)(xrow + k0 + kb);
        float4 xa1 = *(const float4*)(xrow + k0 + kb + 4);
        float4 xa2 = *(const float4*)(xrow + k0 + kb + 8);
        float4 xa3 = *(const float4*)(xrow + k0 + kb + 12);
        ABfrag A;
        A.u[0] = packbf2(xa0.x, xa0.y); A.u[1] = packbf2(xa0.z, xa0.w);
        A.u[2] = packbf2(xa1.x, xa1.y); A.u[3] = packbf2(xa1.z, xa1.w);
        A.u[4] = packbf2(xa2.x, xa2.y); A.u[5] = packbf2(xa2.z, xa2.w);
        A.u[6] = packbf2(xa3.x, xa3.y); A.u[7] = packbf2(xa3.z, xa3.w);

        acc0 = __builtin_amdgcn_wmma_f32_16x16x32_bf16(false, A.v, false, B0.v,
                                                       (short)0, acc0, false, false);
        acc1 = __builtin_amdgcn_wmma_f32_16x16x32_bf16(false, A.v, false, B1.v,
                                                       (short)0, acc1, false, false);
        acc2 = __builtin_amdgcn_wmma_f32_16x16x32_bf16(false, A.v, false, B2.v,
                                                       (short)0, acc2, false, false);
        acc3 = __builtin_amdgcn_wmma_f32_16x16x32_bf16(false, A.v, false, B3.v,
                                                       (short)0, acc3, false, false);
    }

    #pragma unroll
    for (int nt = 0; nt < 4; ++nt) {
        const v8f acc = (nt == 0) ? acc0 : (nt == 1) ? acc1 : (nt == 2) ? acc2 : acc3;
        const int col = nt * 16 + lm;
        const int row0 = rowbase + 8 * lhi;
        #pragma unroll
        for (int r = 0; r < 8; ++r)
            Wh[(size_t)(row0 + r) * F_OUT + col] = acc[r];
        uint4 pk;
        pk.x = packbf2(acc[0], acc[1]);
        pk.y = packbf2(acc[2], acc[3]);
        pk.z = packbf2(acc[4], acc[5]);
        pk.w = packbf2(acc[6], acc[7]);
        *(uint4*)(WhT + (size_t)col * N_NODES + row0) = pk;
    }
}

// ---------------------------------------------------------------------------
// Kernel 2: s1[i] = Wh[i]·a[0:64], s2[i] = Wh[i]·a[64:128]
// ---------------------------------------------------------------------------
__global__ void gat_score_kernel(const float* __restrict__ Wh,
                                 const float* __restrict__ a,
                                 float* __restrict__ s1,
                                 float* __restrict__ s2) {
    int row = blockIdx.x * blockDim.x + threadIdx.x;
    const float4* wr = (const float4*)(Wh + (size_t)row * F_OUT);
    const float4* av = (const float4*)a;
    float acc1 = 0.f, acc2 = 0.f;
    #pragma unroll
    for (int j = 0; j < 16; ++j) {
        float4 w = wr[j];
        float4 a1 = av[j];
        float4 a2 = av[j + 16];
        acc1 += w.x * a1.x + w.y * a1.y + w.z * a1.z + w.w * a1.w;
        acc2 += w.x * a2.x + w.y * a2.y + w.z * a2.z + w.w * a2.w;
    }
    s1[row] = acc1;
    s2[row] = acc2;
}

// ---------------------------------------------------------------------------
// Kernel 3: fused masked softmax attention, column-chunked flash-style.
// Block = 8 waves x 16-row strips; blockIdx encodes (rowblock, chunk).
// Each wave produces UNNORMALIZED partials over 2048 columns:
//   pacc[chunk][row][64], pm[chunk][row], pl[chunk][row]
// ---------------------------------------------------------------------------
__global__ void __launch_bounds__(256)
gat_attn_kernel(const int* __restrict__ adj,
                const unsigned short* __restrict__ WhT,
                const float* __restrict__ s1,
                const float* __restrict__ s2,
                float* __restrict__ pacc,
                float* __restrict__ pm,
                float* __restrict__ pl) {
    const int chunk  = blockIdx.x & (NCHUNK - 1);
    const int rowblk = blockIdx.x >> 2;
    const int wave = threadIdx.x >> 5;
    const int lane = threadIdx.x & 31;
    const int lm   = lane & 15;
    const int lhi  = lane >> 4;
    const int rowbase = rowblk * 128 + wave * 16;
    const int rowA = rowbase + lm;

    const float s1row = s1[rowA];
    const int* adjrow = adj + (size_t)rowA * N_NODES;

    float mrow = -__builtin_inff();
    float lrow = 0.f;
    v8f acc0 = {}, acc1 = {}, acc2 = {}, acc3 = {};

    const int t0i = chunk * (CHUNK_COLS / 32);
    #pragma unroll 1
    for (int t = t0i; t < t0i + CHUNK_COLS / 32; ++t) {
        const int c0  = t * 32;
        const int ckb = c0 + lhi * 16;

        if (t + 1 < t0i + CHUNK_COLS / 32)
            __builtin_prefetch(adjrow + ckb + 32, 0, 1);

        // B fragments (WhT, L2-resident) + adjacency (non-temporal, streamed)
        ABfrag B0, B1, B2, B3;
        {
            const unsigned short* bp0 = WhT + (size_t)(0 * 16 + lm) * N_NODES + c0 + lhi * 16;
            const unsigned short* bp1 = WhT + (size_t)(1 * 16 + lm) * N_NODES + c0 + lhi * 16;
            const unsigned short* bp2 = WhT + (size_t)(2 * 16 + lm) * N_NODES + c0 + lhi * 16;
            const unsigned short* bp3 = WhT + (size_t)(3 * 16 + lm) * N_NODES + c0 + lhi * 16;
            *(uint4*)&B0.u[0] = *(const uint4*)(bp0);
            *(uint4*)&B0.u[4] = *(const uint4*)(bp0 + 8);
            *(uint4*)&B1.u[0] = *(const uint4*)(bp1);
            *(uint4*)&B1.u[4] = *(const uint4*)(bp1 + 8);
            *(uint4*)&B2.u[0] = *(const uint4*)(bp2);
            *(uint4*)&B2.u[4] = *(const uint4*)(bp2 + 8);
            *(uint4*)&B3.u[0] = *(const uint4*)(bp3);
            *(uint4*)&B3.u[4] = *(const uint4*)(bp3 + 8);
        }
        v4i m0 = __builtin_nontemporal_load((const v4i*)(adjrow + ckb));
        v4i m1 = __builtin_nontemporal_load((const v4i*)(adjrow + ckb + 4));
        v4i m2 = __builtin_nontemporal_load((const v4i*)(adjrow + ckb + 8));
        v4i m3 = __builtin_nontemporal_load((const v4i*)(adjrow + ckb + 12));
        float4 t0 = *(const float4*)(s2 + ckb);
        float4 t1 = *(const float4*)(s2 + ckb + 4);
        float4 t2 = *(const float4*)(s2 + ckb + 8);
        float4 t3 = *(const float4*)(s2 + ckb + 12);

        int   msk[16] = {m0.x, m0.y, m0.z, m0.w, m1.x, m1.y, m1.z, m1.w,
                         m2.x, m2.y, m2.z, m2.w, m3.x, m3.y, m3.z, m3.w};
        float sv[16]  = {t0.x, t0.y, t0.z, t0.w, t1.x, t1.y, t1.z, t1.w,
                         t2.x, t2.y, t2.z, t2.w, t3.x, t3.y, t3.z, t3.w};

        float e[16];
        float tmax = -__builtin_inff();
        #pragma unroll
        for (int j = 0; j < 16; ++j) {
            float ev = s1row + sv[j];
            ev = fmaxf(ev, ALPHA * ev);               // leaky relu
            ev = (msk[j] != 0) ? ev : NEG_BIG;        // adjacency mask
            e[j] = ev;
            tmax = fmaxf(tmax, ev);
        }
        tmax = fmaxf(tmax, __shfl_xor(tmax, 16, 32));
        const float newm = fmaxf(mrow, tmax);

        float p[16];
        float psum = 0.f;
        #pragma unroll
        for (int j = 0; j < 16; ++j) {
            p[j] = __expf(e[j] - newm);
            psum += p[j];
        }
        psum += __shfl_xor(psum, 16, 32);

        // Rescale only when some row max changed (wave-uniform branch).
        if (__any(newm > mrow)) {
            const float scale = __expf(mrow - newm);
            #pragma unroll
            for (int r = 0; r < 8; ++r) {
                const float sc = __shfl(scale, 8 * lhi + r, 32);
                acc0[r] *= sc; acc1[r] *= sc; acc2[r] *= sc; acc3[r] *= sc;
            }
            lrow = lrow * scale + psum;
        } else {
            lrow += psum;
        }
        mrow = newm;

        ABfrag A;
        #pragma unroll
        for (int v = 0; v < 8; ++v)
            A.u[v] = packbf2(p[2 * v], p[2 * v + 1]);

        acc0 = __builtin_amdgcn_wmma_f32_16x16x32_bf16(false, A.v, false, B0.v,
                                                       (short)0, acc0, false, false);
        acc1 = __builtin_amdgcn_wmma_f32_16x16x32_bf16(false, A.v, false, B1.v,
                                                       (short)0, acc1, false, false);
        acc2 = __builtin_amdgcn_wmma_f32_16x16x32_bf16(false, A.v, false, B2.v,
                                                       (short)0, acc2, false, false);
        acc3 = __builtin_amdgcn_wmma_f32_16x16x32_bf16(false, A.v, false, B3.v,
                                                       (short)0, acc3, false, false);
    }

    // store unnormalized partials
    if (lane < 16) {
        pm[chunk * N_NODES + rowA] = mrow;
        pl[chunk * N_NODES + rowA] = lrow;
    }
    float* pacc_c = pacc + (size_t)chunk * N_NODES * F_OUT;
    #pragma unroll
    for (int nt = 0; nt < 4; ++nt) {
        const v8f acc = (nt == 0) ? acc0 : (nt == 1) ? acc1 : (nt == 2) ? acc2 : acc3;
        #pragma unroll
        for (int r = 0; r < 8; ++r) {
            const int row = rowbase + 8 * lhi + r;
            pacc_c[(size_t)row * F_OUT + nt * 16 + lm] = acc[r];
        }
    }
}

// ---------------------------------------------------------------------------
// Kernel 4: merge NCHUNK partials per row: softmax-consistent combine.
// ---------------------------------------------------------------------------
__global__ void gat_merge_kernel(const float* __restrict__ pacc,
                                 const float* __restrict__ pm,
                                 const float* __restrict__ pl,
                                 float* __restrict__ out) {
    int i = blockIdx.x * blockDim.x + threadIdx.x;   // 0 .. N*F_OUT-1
    int row = i >> 6;
    float m[NCHUNK];
    float mstar = -__builtin_inff();
    #pragma unroll
    for (int c = 0; c < NCHUNK; ++c) {
        m[c] = pm[c * N_NODES + row];
        mstar = fmaxf(mstar, m[c]);
    }
    float num = 0.f, den = 0.f;
    #pragma unroll
    for (int c = 0; c < NCHUNK; ++c) {
        const float w = __expf(m[c] - mstar);
        num += pacc[(size_t)c * N_NODES * F_OUT + i] * w;
        den += pl[c * N_NODES + row] * w;
    }
    typedef __attribute__((ext_vector_type(1))) float v1f;
    v1f r; r.x = num / den;
    __builtin_nontemporal_store(r, (v1f*)(out + i));
}

// ---------------------------------------------------------------------------
extern "C" void kernel_launch(void* const* d_in, const int* in_sizes, int n_in,
                              void* d_out, int out_size, void* d_ws, size_t ws_size,
                              hipStream_t stream) {
    const float* x   = (const float*)d_in[0];   // [8192, 512]
    const int*   adj = (const int*)d_in[1];     // [8192, 8192]
    const float* W   = (const float*)d_in[2];   // [512, 64]
    const float* a   = (const float*)d_in[3];   // [128, 1]
    float* out = (float*)d_out;                 // [8192, 64]

    char* ws = (char*)d_ws;
    float*          Wh   = (float*)(ws);                         // 2 MB
    unsigned short* WhT  = (unsigned short*)(ws + 2097152);      // 1 MB
    unsigned short* WT   = (unsigned short*)(ws + 3145728);      // 64 KB
    float*          s1   = (float*)(ws + 3211264);               // 32 KB
    float*          s2   = (float*)(ws + 3244032);               // 32 KB
    float*          pacc = (float*)(ws + 3276800);               // 8 MB (4x8192x64)
    float*          pm   = (float*)(ws + 3276800 + 8388608);     // 128 KB
    float*          pl   = (float*)(ws + 3276800 + 8388608 + 131072); // 128 KB

    gat_wt_kernel<<<(F_IN * F_OUT) / 256, 256, 0, stream>>>(W, WT);
    gat_proj_kernel<<<N_NODES / 128, 256, 0, stream>>>(x, WT, Wh, WhT);
    gat_score_kernel<<<N_NODES / 256, 256, 0, stream>>>(Wh, a, s1, s2);
    gat_attn_kernel<<<(N_NODES / 128) * NCHUNK, 256, 0, stream>>>(
        adj, WhT, s1, s2, pacc, pm, pl);
    gat_merge_kernel<<<(N_NODES * F_OUT) / 256, 256, 0, stream>>>(pacc, pm, pl, out);
}
